// ResidualVectorQuantizationWithClustering_85444079386898
// MI455X (gfx1250) — matrix operand order repaired and checked by reading, not compiled, not run on hardware
//
#include <hip/hip_runtime.h>

// CDNA5 / gfx1250 Residual VQ.
// Strategy: bf16 hi/lo split (Markidis) on v_wmma_f32_16x16x32_bf16.
// scores = 0.5*||c||^2 + S·c  with S = -residual  (argmin-equivalent).
// Residual update: S += c[argmin].  Output: out = features + S_final.
//
// Roofline: 2*N*K*D*L = 275 GFLOP of matmul vs ~268 MB HBM (12 us @ 23.3 TB/s);
// codebooks (4 MB) stay L2-resident. Compute-bound -> bf16 WMMA with 3-term
// split (~fp32-accurate scores) at ~3x bf16 MAC count beats f32 16x16x4 WMMA
// by ~4-5x. LDS tile stride padded to 132 dwords so B-fragment ds_load_b128
// spreads 16 lanes over all 64 banks (stride 128 dwords would be 16-way
// same-bank conflict).

typedef __attribute__((ext_vector_type(16))) __bf16 v16bf;
typedef __attribute__((ext_vector_type(8)))  float  v8f;

#define DIM   256        // feature dim
#define DIMP  (DIM+8)    // padded LDS row: 264 ushorts = 132 dwords (132%64=4)
#define NCLU  1024       // clusters per level
#define NLVL  4          // levels
#define WAVES 8          // waves per workgroup
#define MTILE (WAVES*16) // 128 rows per workgroup
#define CT    16         // clusters per LDS tile (one WMMA N-tile)
#define NT    (NCLU/CT)  // 64 cluster tiles
#define KT    (DIM/32)   // 8 k-tiles of 32 (bf16 WMMA depth)

union BF16x16 { v16bf v; unsigned short s[16]; uint4 q[2]; };

__device__ __forceinline__ unsigned short bf16_rne(float x) {
  unsigned int b = __float_as_uint(x);
  b += 0x7FFFu + ((b >> 16) & 1u);          // round-to-nearest-even to bf16
  return (unsigned short)(b >> 16);
}
__device__ __forceinline__ float bf16f(unsigned short h) {
  return __uint_as_float(((unsigned int)h) << 16);
}
__device__ __forceinline__ void split2(float x, unsigned short &h, unsigned short &l) {
  h = bf16_rne(x);
  l = bf16_rne(x - bf16f(h));               // lo captures next ~8 mantissa bits
}

extern "C" __global__ __launch_bounds__(256, 1)
void rvq_wmma_kernel(const float* __restrict__ feats,
                     const float* __restrict__ cbs,
                     float* __restrict__ out) {
  // Double-buffered codebook tile: hi/lo bf16 (padded rows) + 0.5*||c||^2.
  __shared__ __align__(16) unsigned short cbhi[2][CT][DIMP];
  __shared__ __align__(16) unsigned short cblo[2][CT][DIMP];
  __shared__ float cnh[2][CT];
  __shared__ int   idxs[WAVES][16];

  const int tid  = threadIdx.x;
  const int wave = tid >> 5;
  const int lane = tid & 31;
  const int col  = lane & 15;              // row-in-wave (A) / cluster-col (B,C)
  const int ka   = (lane & 16) ? 8  : 0;   // A-fragment K offset (ISA 7.12.2)
  const int kb   = (lane & 16) ? 16 : 0;   // unused (B fragment uses ka)
  const int sc   = tid >> 4;               // stager: cluster row 0..15
  const int seg  = tid & 15;               // stager: 16-dim segment

  const long rowg = (long)blockIdx.x * MTILE + wave * 16 + col;
  const float* frow = feats + rowg * DIM;

  // ---- S = -features, persistent bf16 hi/lo A-fragments (128 VGPRs) ----
  // A layout: lane<16 holds row=col, K runs [32kt, +8) and [32kt+16, +8);
  //           lane>=16 same row, runs shifted by +8.
  v16bf Shi[KT], Slo[KT];
  #pragma unroll
  for (int kt = 0; kt < KT; ++kt) {
    const float* p = frow + 32*kt + ka;
    float4 a0 = ((const float4*)p)[0];
    float4 a1 = ((const float4*)p)[1];
    float4 b0 = ((const float4*)(p + 16))[0];
    float4 b1 = ((const float4*)(p + 16))[1];
    float r[16] = {a0.x,a0.y,a0.z,a0.w, a1.x,a1.y,a1.z,a1.w,
                   b0.x,b0.y,b0.z,b0.w, b1.x,b1.y,b1.z,b1.w};
    BF16x16 h, l;
    #pragma unroll
    for (int e = 0; e < 16; ++e) split2(-r[e], h.s[e], l.s[e]);
    Shi[kt] = h.v; Slo[kt] = l.v;
  }

  for (int lvl = 0; lvl < NLVL; ++lvl) {
    const float* cbl = cbs + (long)lvl * NCLU * DIM;

    // Cooperative tile stager: 256 threads, each converts 16 f32 -> bf16 hi/lo
    // (packed, 2x ds_store_b128 per array) and reduces 0.5*||c||^2 in-wave.
    float sreg[16];
    auto stageLoad = [&](int t) {
      const float* src = cbl + (long)(t * CT + sc) * DIM + seg * 16;
      #pragma unroll
      for (int j = 0; j < 4; ++j) {
        float4 v = ((const float4*)src)[j];
        sreg[4*j+0]=v.x; sreg[4*j+1]=v.y; sreg[4*j+2]=v.z; sreg[4*j+3]=v.w;
      }
    };
    auto stageStore = [&](int buf) {
      unsigned int hw[8], lw[8];
      float ss = 0.f;
      #pragma unroll
      for (int j = 0; j < 8; ++j) {
        unsigned short h0, l0, h1, l1;
        split2(sreg[2*j+0], h0, l0);
        split2(sreg[2*j+1], h1, l1);
        hw[j] = (unsigned int)h0 | ((unsigned int)h1 << 16);
        lw[j] = (unsigned int)l0 | ((unsigned int)l1 << 16);
        ss += sreg[2*j]*sreg[2*j] + sreg[2*j+1]*sreg[2*j+1];
      }
      uint4* dh = (uint4*)&cbhi[buf][sc][seg * 16];
      dh[0] = make_uint4(hw[0], hw[1], hw[2], hw[3]);
      dh[1] = make_uint4(hw[4], hw[5], hw[6], hw[7]);
      uint4* dl = (uint4*)&cblo[buf][sc][seg * 16];
      dl[0] = make_uint4(lw[0], lw[1], lw[2], lw[3]);
      dl[1] = make_uint4(lw[4], lw[5], lw[6], lw[7]);
      #pragma unroll
      for (int m = 1; m < 16; m <<= 1) ss += __shfl_xor(ss, m, 32);
      if (seg == 0) cnh[buf][sc] = 0.5f * ss;
    };

    float minv[8]; int mini[8];
    #pragma unroll
    for (int i = 0; i < 8; ++i) { minv[i] = 3.402823e38f; mini[i] = 0; }

    stageLoad(0); stageStore(0);
    __syncthreads();

    for (int t = 0; t < NT; ++t) {
      const int  buf  = t & 1;
      const bool more = (t + 1) < NT;
      if (more) stageLoad(t + 1);          // global loads overlap WMMA below

      // Two accumulators (12+12 WMMA chains) summed at the end of the tile.
      v8f acc0, acc1;
      const float ci = cnh[buf][col];
      #pragma unroll
      for (int i = 0; i < 8; ++i) { acc0[i] = ci; acc1[i] = 0.f; }

      #pragma unroll
      for (int kt = 0; kt < KT; ++kt) {
        BF16x16 bh, bl;                    // B: K runs [32kt+ka, +8) and [32kt+16+ka, +8), col=lane%16
        const uint4* ph = (const uint4*)&cbhi[buf][col][32*kt + ka];
        bh.q[0] = ph[0]; bh.q[1] = ph[2];
        const uint4* pl = (const uint4*)&cblo[buf][col][32*kt + ka];
        bl.q[0] = pl[0]; bl.q[1] = pl[2];
        if (kt & 1) {
          acc1 = __builtin_amdgcn_wmma_f32_16x16x32_bf16(false, Shi[kt], false, bh.v,
                                                         (short)0, acc1, false, false);
          acc0 = __builtin_amdgcn_wmma_f32_16x16x32_bf16(false, Shi[kt], false, bl.v,
                                                         (short)0, acc0, false, false);
          acc1 = __builtin_amdgcn_wmma_f32_16x16x32_bf16(false, Slo[kt], false, bh.v,
                                                         (short)0, acc1, false, false);
        } else {
          acc0 = __builtin_amdgcn_wmma_f32_16x16x32_bf16(false, Shi[kt], false, bh.v,
                                                         (short)0, acc0, false, false);
          acc1 = __builtin_amdgcn_wmma_f32_16x16x32_bf16(false, Shi[kt], false, bl.v,
                                                         (short)0, acc1, false, false);
          acc0 = __builtin_amdgcn_wmma_f32_16x16x32_bf16(false, Slo[kt], false, bh.v,
                                                         (short)0, acc0, false, false);
        }
      }

      const int cbase = t * CT + col;
      #pragma unroll
      for (int i = 0; i < 8; ++i) {
        const float s = acc0[i] + acc1[i];
        if (s < minv[i]) { minv[i] = s; mini[i] = cbase; }
      }

      if (more) stageStore(buf ^ 1);
      __syncthreads();
    }

    // Butterfly argmin over the 16 lanes sharing each row; ties -> lowest idx
    // (matches jnp.argmin first-occurrence).
    #pragma unroll
    for (int i = 0; i < 8; ++i) {
      float v = minv[i]; int ix = mini[i];
      #pragma unroll
      for (int m = 1; m < 16; m <<= 1) {
        float v2 = __shfl_xor(v, m, 32);
        int   i2 = __shfl_xor(ix, m, 32);
        if (v2 < v || (v2 == v && i2 < ix)) { v = v2; ix = i2; }
      }
      mini[i] = ix;
    }
    if (lane == 0) {
      #pragma unroll
      for (int i = 0; i < 8; ++i) idxs[wave][i] = mini[i];      // rows 0..7
    } else if (lane == 16) {
      #pragma unroll
      for (int i = 0; i < 8; ++i) idxs[wave][8 + i] = mini[i];  // rows 8..15
    }
    const int myidx = idxs[wave][col];   // same-wave LDS RAW; compiler waits DScnt

    // Residual update: S += c[myidx], re-split to hi/lo.
    const float* qrow = cbl + (long)myidx * DIM;
    #pragma unroll
    for (int kt = 0; kt < KT; ++kt) {
      const float* p = qrow + 32*kt + ka;
      float4 a0 = ((const float4*)p)[0];
      float4 a1 = ((const float4*)p)[1];
      float4 b0 = ((const float4*)(p + 16))[0];
      float4 b1 = ((const float4*)(p + 16))[1];
      float q[16] = {a0.x,a0.y,a0.z,a0.w, a1.x,a1.y,a1.z,a1.w,
                     b0.x,b0.y,b0.z,b0.w, b1.x,b1.y,b1.z,b1.w};
      BF16x16 oh, ol, nh, nl;
      oh.v = Shi[kt]; ol.v = Slo[kt];
      #pragma unroll
      for (int e = 0; e < 16; ++e) {
        float s = bf16f(oh.s[e]) + bf16f(ol.s[e]) + q[e];
        split2(s, nh.s[e], nl.s[e]);
      }
      Shi[kt] = nh.v; Slo[kt] = nl.v;
    }
    __syncthreads();   // keep level phases aligned before next staging pass
  }

  // out = features + S_final  (== sum of selected centers)
  float* orow = out + rowg * DIM;
  #pragma unroll
  for (int kt = 0; kt < KT; ++kt) {
    const float* p = frow + 32*kt + ka;
    float4 a0 = ((const float4*)p)[0];
    float4 a1 = ((const float4*)p)[1];
    float4 b0 = ((const float4*)(p + 16))[0];
    float4 b1 = ((const float4*)(p + 16))[1];
    float f[16] = {a0.x,a0.y,a0.z,a0.w, a1.x,a1.y,a1.z,a1.w,
                   b0.x,b0.y,b0.z,b0.w, b1.x,b1.y,b1.z,b1.w};
    BF16x16 oh, ol; oh.v = Shi[kt]; ol.v = Slo[kt];
    float o[16];
    #pragma unroll
    for (int e = 0; e < 16; ++e) o[e] = f[e] + bf16f(oh.s[e]) + bf16f(ol.s[e]);
    float* d = orow + 32*kt + ka;
    ((float4*)d)[0]        = make_float4(o[0],  o[1],  o[2],  o[3]);
    ((float4*)d)[1]        = make_float4(o[4],  o[5],  o[6],  o[7]);
    ((float4*)(d + 16))[0] = make_float4(o[8],  o[9],  o[10], o[11]);
    ((float4*)(d + 16))[1] = make_float4(o[12], o[13], o[14], o[15]);
  }
}

extern "C" void kernel_launch(void* const* d_in, const int* in_sizes, int n_in,
                              void* d_out, int out_size, void* d_ws, size_t ws_size,
                              hipStream_t stream) {
  const float* feats = (const float*)d_in[0];   // [N, 256] f32
  const float* cbs   = (const float*)d_in[1];   // [4, 1024, 256] f32
  float* outp        = (float*)d_out;           // [N, 256] f32
  const int n    = in_sizes[0] / DIM;           // 131072
  const int grid = n / MTILE;                   // 1024 workgroups of 256 threads
  rvq_wmma_kernel<<<grid, WAVES * 32, 0, stream>>>(feats, cbs, outp);
}
